// RNN_73710228734663
// MI455X (gfx1250) — compile-verified, matrix operand order checked by reading, not run on hardware
//
#include <hip/hip_runtime.h>
#include <hip/hip_bf16.h>

typedef __attribute__((ext_vector_type(16))) _Float16 v16h;
typedef __attribute__((ext_vector_type(8)))  float    v8f;

#define Bsz   512
#define Ssz   128
#define NEc   8192
#define EDc   128
#define DDc   32
#define Hh    512
#define FourH 2048
#define XF    160          // ED + DD
#define Kc    672          // XF + H
#define NDc   8
#define PLc   5
#define SIMN  40960        // NE * PL

// ---------------- WMMA helpers ----------------

// Fragment loader for both A (16xK tile, rows) and B (Kx16 tile, via row-major
// "weight" matrix whose row n is column n of B). ISA 16-bit layout:
// lane l: row/col = l&15, kbase = (l>>4)*8; elems [0..7] at k=kbase+0..7,
// elems [8..15] at k=kbase+16..23.  Two 16B contiguous loads per lane.
__device__ __forceinline__ v16h load_frag(const _Float16* __restrict__ p0,
                                          int stride, int lane) {
  const _Float16* p = p0 + (size_t)(lane & 15) * stride + ((lane >> 4) * 8);
  v16h f;
#pragma unroll
  for (int i = 0; i < 8; ++i) f[i] = p[i];
#pragma unroll
  for (int i = 0; i < 8; ++i) f[8 + i] = p[16 + i];
  return f;
}

__device__ __forceinline__ v8f wmma16(v16h a, v16h b, v8f c) {
  return __builtin_amdgcn_wmma_f32_16x16x32_f16(
      /*neg_a=*/false, a, /*neg_b=*/false, b,
      /*c_mod=*/(short)0, c, /*reuse_a=*/false, /*reuse_b=*/false);
}

__device__ __forceinline__ float sigmoidf_(float x) {
  return 1.0f / (1.0f + __expf(-x));
}

// ---------------- prep kernels ----------------

__global__ void k_convert_f16(const float* __restrict__ src,
                              _Float16* __restrict__ dst, int n) {
  int t = blockIdx.x * 256 + threadIdx.x;
  if (t < n) dst[t] = (_Float16)src[t];
}

__global__ void k_prep_wc(const float* __restrict__ w_ih,
                          const float* __restrict__ w_hh,
                          _Float16* __restrict__ Wc) {
  int t = blockIdx.x * 256 + threadIdx.x;          // 2048*672
  if (t >= FourH * Kc) return;
  int n = t / Kc, k = t % Kc;
  float v = (k < XF) ? w_ih[n * XF + k] : w_hh[n * Hh + (k - XF)];
  Wc[t] = (_Float16)v;
}

__global__ void k_prep_bias(const float* __restrict__ b_ih,
                            const float* __restrict__ b_hh,
                            float* __restrict__ bc) {
  int t = blockIdx.x * 256 + threadIdx.x;
  if (t < FourH) bc[t] = b_ih[t] + b_hh[t];
}

// x = concat(link_emb[inputs], dir_emb[directions]) stored as (S, B, 160) f16
__global__ void k_embed(const int* __restrict__ inputs,
                        const int* __restrict__ dirs,
                        const float* __restrict__ link_w,
                        const float* __restrict__ dir_w,
                        _Float16* __restrict__ X) {
  size_t t = (size_t)blockIdx.x * 256 + threadIdx.x;   // S*B*160
  if (t >= (size_t)Ssz * Bsz * XF) return;
  int k = (int)(t % XF);
  size_t sb = t / XF;
  int b = (int)(sb % Bsz), s = (int)(sb / Bsz);
  float v;
  if (k < EDc) v = link_w[(size_t)inputs[b * Ssz + s] * EDc + k];
  else         v = dir_w[(size_t)dirs[b * Ssz + s] * DDc + (k - EDc)];
  X[t] = (_Float16)v;
}

__global__ void k_zero_state(float* __restrict__ c, _Float16* __restrict__ hf16) {
  int t = blockIdx.x * 256 + threadIdx.x;              // B*H
  if (t < Bsz * Hh) { c[t] = 0.0f; hf16[t] = (_Float16)0.0f; }
}

// ---------------- LSTM step: gates GEMM ----------------
// gates(B,2048) = [x_t | h](B,672) @ Wc(2048,672)^T + biasc
// Register-blocked: grid (8, 16), block 128 (4 waves).
// Each wave: 4 M-tiles x 2 N-tiles = 8 accumulators; per K-step all
// 6 fragment loads issue first, then 8 WMMAs (loads of kt+1 hide under
// WMMAs of kt).
__global__ void k_gates_gemm(const _Float16* __restrict__ Xf16,
                             const _Float16* __restrict__ hf16,
                             const _Float16* __restrict__ Wc,
                             const float* __restrict__ biasc,
                             float* __restrict__ gates, int t) {
  int lane = threadIdx.x & 31;
  int wave = threadIdx.x >> 5;
  int row0 = blockIdx.x * 64;
  int n0 = (blockIdx.y * 8 + wave * 2) * 16;
  int half = lane >> 4, lcol = lane & 15;
  float b0 = biasc[n0 + lcol], b1 = biasc[n0 + 16 + lcol];
  v8f acc0[4], acc1[4];
#pragma unroll
  for (int m = 0; m < 4; ++m) {
    acc0[m] = (v8f){b0, b0, b0, b0, b0, b0, b0, b0};
    acc1[m] = (v8f){b1, b1, b1, b1, b1, b1, b1, b1};
  }
  const _Float16* xbase = Xf16 + ((size_t)t * Bsz + row0) * XF;
  const _Float16* hbase = hf16 + (size_t)row0 * Hh;
  const _Float16* wb0 = Wc + (size_t)n0 * Kc;
  const _Float16* wb1 = Wc + (size_t)(n0 + 16) * Kc;
#pragma unroll
  for (int kt = 0; kt < Kc / 32; ++kt) {
    v16h bb0 = load_frag(wb0 + kt * 32, Kc, lane);
    v16h bb1 = load_frag(wb1 + kt * 32, Kc, lane);
    v16h a[4];
#pragma unroll
    for (int m = 0; m < 4; ++m) {
      a[m] = (kt < 5)
                 ? load_frag(xbase + (size_t)(m * 16) * XF + kt * 32, XF, lane)
                 : load_frag(hbase + (size_t)(m * 16) * Hh + (kt - 5) * 32, Hh, lane);
    }
#pragma unroll
    for (int m = 0; m < 4; ++m) {
      acc0[m] = wmma16(a[m], bb0, acc0[m]);
      acc1[m] = wmma16(a[m], bb1, acc1[m]);
    }
  }
#pragma unroll
  for (int m = 0; m < 4; ++m) {
#pragma unroll
    for (int v = 0; v < 8; ++v) {
      int row = row0 + m * 16 + v + 8 * half;
      gates[(size_t)row * FourH + n0 + lcol]      = acc0[m][v];
      gates[(size_t)row * FourH + n0 + 16 + lcol] = acc1[m][v];
    }
  }
}

// ---------------- LSTM elementwise cell ----------------
__global__ void k_lstm_cell(const float* __restrict__ gates,
                            float* __restrict__ c,
                            _Float16* __restrict__ hf16) {
  int idx = blockIdx.x * 256 + threadIdx.x;            // B*H
  if (idx >= Bsz * Hh) return;
  int b = idx >> 9, j = idx & 511;
  const float* g = gates + (size_t)b * FourH;
  float ig = sigmoidf_(g[j]);
  float fg = sigmoidf_(g[Hh + j]);
  float gg = tanhf(g[2 * Hh + j]);
  float og = sigmoidf_(g[3 * Hh + j]);
  float cn = fg * c[idx] + ig * gg;
  c[idx] = cn;
  hf16[idx] = (_Float16)(og * tanhf(cn));
}

// ---------------- MLP relu GEMM: out(512,N) = relu(A(512,K) @ W(N,K)^T + b) ----------------
// grid (8, N/128), block 128; 4 M-tiles x 2 N-tiles per wave
__global__ void k_gemm_relu(const _Float16* __restrict__ A,
                            const _Float16* __restrict__ W,
                            const float* __restrict__ bias,
                            _Float16* __restrict__ out, int K, int N) {
  int lane = threadIdx.x & 31;
  int wave = threadIdx.x >> 5;
  int row0 = blockIdx.x * 64;
  int n0 = (blockIdx.y * 8 + wave * 2) * 16;
  int half = lane >> 4, lcol = lane & 15;
  float b0 = bias[n0 + lcol], b1 = bias[n0 + 16 + lcol];
  v8f acc0[4], acc1[4];
#pragma unroll
  for (int m = 0; m < 4; ++m) {
    acc0[m] = (v8f){b0, b0, b0, b0, b0, b0, b0, b0};
    acc1[m] = (v8f){b1, b1, b1, b1, b1, b1, b1, b1};
  }
  const _Float16* ab = A + (size_t)row0 * K;
  const _Float16* wb0 = W + (size_t)n0 * K;
  const _Float16* wb1 = W + (size_t)(n0 + 16) * K;
  for (int kt = 0; kt < K / 32; ++kt) {
    v16h bb0 = load_frag(wb0 + kt * 32, K, lane);
    v16h bb1 = load_frag(wb1 + kt * 32, K, lane);
    v16h a[4];
#pragma unroll
    for (int m = 0; m < 4; ++m)
      a[m] = load_frag(ab + (size_t)(m * 16) * K + kt * 32, K, lane);
#pragma unroll
    for (int m = 0; m < 4; ++m) {
      acc0[m] = wmma16(a[m], bb0, acc0[m]);
      acc1[m] = wmma16(a[m], bb1, acc1[m]);
    }
  }
#pragma unroll
  for (int m = 0; m < 4; ++m) {
#pragma unroll
    for (int v = 0; v < 8; ++v) {
      int row = row0 + m * 16 + v + 8 * half;
      out[(size_t)row * N + n0 + lcol]      = (_Float16)fmaxf(acc0[m][v], 0.0f);
      out[(size_t)row * N + n0 + 16 + lcol] = (_Float16)fmaxf(acc1[m][v], 0.0f);
    }
  }
}

// ---------------- logits (512 x 8), K=512, scalar fp32 ----------------
__global__ void k_logits(const _Float16* __restrict__ z2,
                         const float* __restrict__ W3,
                         const float* __restrict__ b3,
                         float* __restrict__ logits) {
  int t = blockIdx.x * 256 + threadIdx.x;              // 512*8
  if (t >= Bsz * NDc) return;
  int b = t >> 3, d = t & 7;
  float s = b3[d];
  const _Float16* z = z2 + (size_t)b * Hh;
  const float* w = W3 + (size_t)d * Hh;
  for (int k = 0; k < Hh; ++k) s += (float)z[k] * w[k];
  logits[t] = s;
}

// ---------------- log_softmax^2, loss, top-2, direction_correct ----------------
// single block, 512 threads (one row each)
__global__ void k_loss_topk(const float* __restrict__ logits,
                            const int* __restrict__ inputs,
                            const int* __restrict__ goal,
                            const int* __restrict__ ldm,
                            int* __restrict__ idx_out,
                            float* __restrict__ d_loss,
                            float* __restrict__ d_dc) {
  __shared__ float sl[512];
  __shared__ int   si[512];
  int b = threadIdx.x;
  float x[NDc], p[NDc];
#pragma unroll
  for (int k = 0; k < NDc; ++k) x[k] = logits[b * NDc + k];
  float m = x[0];
#pragma unroll
  for (int k = 1; k < NDc; ++k) m = fmaxf(m, x[k]);
  float s = 0.0f;
#pragma unroll
  for (int k = 0; k < NDc; ++k) s += __expf(x[k] - m);
  float lse = __logf(s);
#pragma unroll
  for (int k = 0; k < NDc; ++k) p[k] = x[k] - m - lse;
  // top-2 (first-index tie-breaking, as jax.lax.top_k)
  int i0 = 0;
#pragma unroll
  for (int k = 1; k < NDc; ++k) if (p[k] > p[i0]) i0 = k;
  int i1 = (i0 == 0) ? 1 : 0;
#pragma unroll
  for (int k = 0; k < NDc; ++k) if (k != i0 && p[k] > p[i1]) i1 = k;
  idx_out[2 * b] = i0;
  idx_out[2 * b + 1] = i1;
  // second log_softmax
  float m2 = p[0];
#pragma unroll
  for (int k = 1; k < NDc; ++k) m2 = fmaxf(m2, p[k]);
  float s2 = 0.0f;
#pragma unroll
  for (int k = 0; k < NDc; ++k) s2 += __expf(p[k] - m2);
  float lse2 = __logf(s2);
  int last = inputs[b * Ssz + (Ssz - 1)];
  int lbl = ldm[(size_t)(last - 1) * NEc + goal[b]];
  sl[b] = -(p[lbl] - m2 - lse2);
  si[b] = (i0 == lbl || i1 == lbl) ? 1 : 0;
  __syncthreads();
  for (int st = 256; st > 0; st >>= 1) {
    if (b < st) { sl[b] += sl[b + st]; si[b] += si[b + st]; }
    __syncthreads();
  }
  if (b == 0) {
    *d_loss = sl[0] * (5.0f / (float)Bsz);
    *d_dc = (float)si[0];
  }
}

// ---------------- query = link_emb[last_obs] + pad(0.5*(de0+de1)) ----------------
__global__ void k_build_query(const int* __restrict__ inputs,
                              const int* __restrict__ idx,
                              const float* __restrict__ link_w,
                              const float* __restrict__ dir_w,
                              _Float16* __restrict__ q) {
  int t = blockIdx.x * 256 + threadIdx.x;              // 512*128
  if (t >= Bsz * EDc) return;
  int b = t >> 7, k = t & 127;
  int last = inputs[b * Ssz + (Ssz - 1)];
  float v = link_w[(size_t)last * EDc + k];
  if (k < DDc) {
    int i0 = idx[2 * b], i1 = idx[2 * b + 1];
    v += 0.5f * (dir_w[(i0 + 1) * DDc + k] + dir_w[(i1 + 1) * DDc + k]);
  }
  q[t] = (_Float16)v;
}

// ---------------- sim GEMM (512 x 8192, K=128) + 5-way tiled store ----------------
// grid (8, 64), block 128; 4 M-tiles x 2 N-tiles per wave
__global__ void k_sim_gemm(const _Float16* __restrict__ Q,
                           const _Float16* __restrict__ L,
                           float* __restrict__ out) {
  int lane = threadIdx.x & 31;
  int wave = threadIdx.x >> 5;
  int row0 = blockIdx.x * 64;
  int n0 = (blockIdx.y * 8 + wave * 2) * 16;
  int half = lane >> 4, lcol = lane & 15;
  v8f acc0[4], acc1[4];
#pragma unroll
  for (int m = 0; m < 4; ++m) {
    acc0[m] = (v8f){0, 0, 0, 0, 0, 0, 0, 0};
    acc1[m] = (v8f){0, 0, 0, 0, 0, 0, 0, 0};
  }
  const _Float16* qb = Q + (size_t)row0 * EDc;
  const _Float16* lb0 = L + (size_t)n0 * EDc;
  const _Float16* lb1 = L + (size_t)(n0 + 16) * EDc;
#pragma unroll
  for (int kt = 0; kt < EDc / 32; ++kt) {
    v16h bb0 = load_frag(lb0 + kt * 32, EDc, lane);
    v16h bb1 = load_frag(lb1 + kt * 32, EDc, lane);
    v16h a[4];
#pragma unroll
    for (int m = 0; m < 4; ++m)
      a[m] = load_frag(qb + (size_t)(m * 16) * EDc + kt * 32, EDc, lane);
#pragma unroll
    for (int m = 0; m < 4; ++m) {
      acc0[m] = wmma16(a[m], bb0, acc0[m]);
      acc1[m] = wmma16(a[m], bb1, acc1[m]);
    }
  }
#pragma unroll
  for (int m = 0; m < 4; ++m) {
#pragma unroll
    for (int v = 0; v < 8; ++v) {
      size_t row = (size_t)(row0 + m * 16 + v + 8 * half) * SIMN;
#pragma unroll
      for (int r = 0; r < PLc; ++r) {
        out[row + (size_t)r * NEc + n0 + lcol]      = acc0[m][v];
        out[row + (size_t)r * NEc + n0 + 16 + lcol] = acc1[m][v];
      }
    }
  }
}

__global__ void k_copy_f32(const float* __restrict__ src,
                           float* __restrict__ dst, int n) {
  int t = blockIdx.x * 256 + threadIdx.x;
  if (t < n) dst[t] = src[t];
}

// ---------------- launch ----------------

extern "C" void kernel_launch(void* const* d_in, const int* in_sizes, int n_in,
                              void* d_out, int out_size, void* d_ws, size_t ws_size,
                              hipStream_t stream) {
  const int*   inputs = (const int*)d_in[0];
  const int*   dirs   = (const int*)d_in[1];
  const int*   goal   = (const int*)d_in[2];
  const int*   ldm    = (const int*)d_in[3];
  const float* pdr    = (const float*)d_in[4];
  const float* link_w = (const float*)d_in[5];
  const float* dir_w  = (const float*)d_in[6];
  const float* w_ih   = (const float*)d_in[7];
  const float* b_ih   = (const float*)d_in[8];
  const float* w_hh   = (const float*)d_in[9];
  const float* b_hh   = (const float*)d_in[10];
  const float* W1     = (const float*)d_in[11];
  const float* b1     = (const float*)d_in[12];
  const float* W2     = (const float*)d_in[13];
  const float* b2     = (const float*)d_in[14];
  const float* W3     = (const float*)d_in[15];
  const float* b3     = (const float*)d_in[16];

  // workspace layout (all offsets 16B-aligned)
  char* w = (char*)d_ws;
  size_t off = 0;
  _Float16* Xf16  = (_Float16*)(w + off); off += (size_t)Ssz * Bsz * XF * 2;   // 20,971,520
  _Float16* Wc    = (_Float16*)(w + off); off += (size_t)FourH * Kc * 2;       //  2,752,512
  _Float16* Lf16  = (_Float16*)(w + off); off += (size_t)NEc * EDc * 2;        //  2,097,152
  _Float16* W1f   = (_Float16*)(w + off); off += (size_t)Hh * Hh * 2;
  _Float16* W2f   = (_Float16*)(w + off); off += (size_t)Hh * Hh * 2;
  _Float16* hf16  = (_Float16*)(w + off); off += (size_t)Bsz * Hh * 2;
  _Float16* z1f   = (_Float16*)(w + off); off += (size_t)Bsz * Hh * 2;
  _Float16* z2f   = (_Float16*)(w + off); off += (size_t)Bsz * Hh * 2;
  _Float16* qf    = (_Float16*)(w + off); off += (size_t)Bsz * EDc * 2;
  float* biasc    = (float*)(w + off);    off += (size_t)FourH * 4;
  float* cbuf     = (float*)(w + off);    off += (size_t)Bsz * Hh * 4;
  float* gates    = (float*)(w + off);    off += (size_t)Bsz * FourH * 4;
  float* logits   = (float*)(w + off);    off += (size_t)Bsz * NDc * 4;
  int*   idxb     = (int*)(w + off);      off += (size_t)Bsz * 2 * 4;

  float* out_ph   = (float*)d_out;                                  // (512, 40960)
  float* out_pdr  = out_ph + (size_t)Bsz * SIMN;                    // (512, 40)
  float* out_loss = out_pdr + (size_t)Bsz * NDc * PLc;              // scalar
  float* out_dc   = out_loss + 1;                                   // scalar

  // --- prep ---
  k_prep_wc<<<(FourH * Kc + 255) / 256, 256, 0, stream>>>(w_ih, w_hh, Wc);
  k_prep_bias<<<(FourH + 255) / 256, 256, 0, stream>>>(b_ih, b_hh, biasc);
  k_convert_f16<<<(Hh * Hh + 255) / 256, 256, 0, stream>>>(W1, W1f, Hh * Hh);
  k_convert_f16<<<(Hh * Hh + 255) / 256, 256, 0, stream>>>(W2, W2f, Hh * Hh);
  // link_emb rows 1..8192 as f16 for sim GEMM
  k_convert_f16<<<(NEc * EDc + 255) / 256, 256, 0, stream>>>(link_w + EDc, Lf16, NEc * EDc);
  {
    size_t n = (size_t)Ssz * Bsz * XF;
    k_embed<<<(unsigned)((n + 255) / 256), 256, 0, stream>>>(inputs, dirs, link_w, dir_w, Xf16);
  }
  k_zero_state<<<(Bsz * Hh + 255) / 256, 256, 0, stream>>>(cbuf, hf16);

  // --- sequential LSTM ---
  dim3 ggrid(Bsz / 64, FourH / 128);
  for (int t = 0; t < Ssz; ++t) {
    k_gates_gemm<<<ggrid, 128, 0, stream>>>(Xf16, hf16, Wc, biasc, gates, t);
    k_lstm_cell<<<(Bsz * Hh + 255) / 256, 256, 0, stream>>>(gates, cbuf, hf16);
  }

  // --- MLP head ---
  dim3 mgrid(Bsz / 64, Hh / 128);
  k_gemm_relu<<<mgrid, 128, 0, stream>>>(hf16, W1f, b1, z1f, Hh, Hh);
  k_gemm_relu<<<mgrid, 128, 0, stream>>>(z1f, W2f, b2, z2f, Hh, Hh);
  k_logits<<<(Bsz * NDc + 255) / 256, 256, 0, stream>>>(z2f, W3, b3, logits);

  // --- loss / top-2 ---
  k_loss_topk<<<1, 512, 0, stream>>>(logits, inputs, goal, ldm, idxb, out_loss, out_dc);

  // --- query + sim + tiled output ---
  k_build_query<<<(Bsz * EDc + 255) / 256, 256, 0, stream>>>(inputs, idxb, link_w, dir_w, qf);
  dim3 sgrid(Bsz / 64, NEc / 128);
  k_sim_gemm<<<sgrid, 128, 0, stream>>>(qf, Lf16, out_ph);

  // --- pred_d_rand passthrough ---
  k_copy_f32<<<(Bsz * NDc * PLc + 255) / 256, 256, 0, stream>>>(pdr, out_pdr, Bsz * NDc * PLc);
}
